// MoELayer_29257317220861
// MI455X (gfx1250) — matrix operand order repaired and checked by reading, not compiled, not run on hardware
//
#include <hip/hip_runtime.h>
#include <hip/hip_bf16.h>

#define N_TOK 16384
#define DDIM  512
#define NEXP  16

typedef __attribute__((ext_vector_type(16))) __bf16 v16bf;
typedef __attribute__((ext_vector_type(8)))  __bf16 v8bf;
typedef __attribute__((ext_vector_type(8)))  float  v8f;
typedef __attribute__((ext_vector_type(4)))  float  f32x4;

// f32 -> bf16 with round-to-nearest-even (used only in the one-pass converter)
static __device__ __forceinline__ __bf16 to_bf16(float f) {
    unsigned u = __builtin_bit_cast(unsigned, f);
    u += 0x7FFFu + ((u >> 16) & 1u);
    unsigned short h = (unsigned short)(u >> 16);
    return __builtin_bit_cast(__bf16, h);
}

// ---------------------------------------------------------------------------
// Kernel 0: one-pass f32 -> bf16 conversion (memory-bound, 8 elems/thread).
// ---------------------------------------------------------------------------
__global__ void __launch_bounds__(256)
convert_bf16_kernel(const float* __restrict__ src, __bf16* __restrict__ dst, int n8)
{
    const int t = blockIdx.x * 256 + threadIdx.x;
    if (t >= n8) return;
    const f32x4* p = (const f32x4*)(src + (size_t)t * 8);
    f32x4 a = p[0], b = p[1];
    v8bf r;
    r[0] = to_bf16(a[0]); r[1] = to_bf16(a[1]); r[2] = to_bf16(a[2]); r[3] = to_bf16(a[3]);
    r[4] = to_bf16(b[0]); r[5] = to_bf16(b[1]); r[6] = to_bf16(b[2]); r[7] = to_bf16(b[3]);
    *(v8bf*)(dst + (size_t)t * 8) = r;
}

// ---------------------------------------------------------------------------
// Kernel 1: gating. One thread per token. Wg staged in LDS (32KB).
// Appends (token | slot<<20, weight) to per-expert lists via atomics.
// ---------------------------------------------------------------------------
__global__ void __launch_bounds__(256)
gate_kernel(const float* __restrict__ x, const float* __restrict__ Wg,
            const float* __restrict__ bg, const float* __restrict__ gb,
            int* __restrict__ cnt, int* __restrict__ list_tok,
            float* __restrict__ list_w)
{
    __shared__ float sWg[NEXP * DDIM];
    for (int i = threadIdx.x; i < NEXP * DDIM; i += 256) sWg[i] = Wg[i];
    __syncthreads();

    const int n = blockIdx.x * 256 + threadIdx.x;
    float acc[NEXP];
    #pragma unroll
    for (int e = 0; e < NEXP; ++e) acc[e] = bg[e] + gb[e];

    const float* xr = x + (size_t)n * DDIM;
    for (int d = 0; d < DDIM; d += 4) {
        f32x4 xv = *(const f32x4*)(xr + d);
        #pragma unroll
        for (int e = 0; e < NEXP; ++e) {
            acc[e] += xv[0] * sWg[e * DDIM + d]     + xv[1] * sWg[e * DDIM + d + 1]
                    + xv[2] * sWg[e * DDIM + d + 2] + xv[3] * sWg[e * DDIM + d + 3];
        }
    }

    // top-2 (ties -> lowest index, matching jax.lax.top_k)
    int i0 = 0; float v0 = acc[0];
    #pragma unroll
    for (int e = 1; e < NEXP; ++e) if (acc[e] > v0) { v0 = acc[e]; i0 = e; }
    int i1 = -1; float v1 = -3.4e38f;
    #pragma unroll
    for (int e = 0; e < NEXP; ++e) if (e != i0 && acc[e] > v1) { v1 = acc[e]; i1 = e; }

    float ex  = __expf(v1 - v0);        // v1 <= v0, stable
    float inv = 1.0f / (1.0f + ex);
    float w0  = inv, w1 = ex * inv;

    int p0 = atomicAdd(&cnt[i0], 1);
    list_tok[i0 * N_TOK + p0] = n;                  // slot 0
    list_w  [i0 * N_TOK + p0] = w0;
    int p1 = atomicAdd(&cnt[i1], 1);
    list_tok[i1 * N_TOK + p1] = n | (1 << 20);      // slot 1
    list_w  [i1 * N_TOK + p1] = w1;
}

// Build the A fragment: lane holds row m; K = {hi*16..hi*16+15}
static __device__ __forceinline__ v16bf load_a(const __bf16* __restrict__ p) {
    v8bf lo = *(const v8bf*)p;
    v8bf hi = *(const v8bf*)(p + 8);
    return __builtin_shufflevector(lo, hi, 0, 1, 2, 3, 4, 5, 6, 7,
                                           8, 9, 10, 11, 12, 13, 14, 15);
}

// ---------------------------------------------------------------------------
// Kernel 2: shared expert GEMM. One wave per 16-token x 128-output tile.
// All 8 waves of a block share the same o-quad (same B stream -> cache hits).
// out[n][o] = sum_d x[n][d] * Ws[o][d] + bs[o]   (x added later in combine)
// ---------------------------------------------------------------------------
__global__ void __launch_bounds__(256)
shared_gemm_kernel(const __bf16* __restrict__ xb, const __bf16* __restrict__ Wsb,
                   const float* __restrict__ bs, float* __restrict__ out)
{
    const int lane  = threadIdx.x & 31;
    const int wave  = threadIdx.x >> 5;
    const int nBase = ((blockIdx.x >> 2) * 8 + wave) * 16;   // 1024 n-tiles
    const int oBase = (blockIdx.x & 3) * 128;                // 4 o-quads
    const int m  = lane & 15;
    const int hi = lane >> 4;

    v8f acc[8] = {};
    const __bf16* xrow = xb + (size_t)(nBase + m) * DDIM;
    const __bf16* wcol = Wsb + (size_t)(oBase + m) * DDIM + hi * 16;

    v16bf a = load_a(xrow + hi * 16);
    #pragma unroll 1
    for (int k0 = 0; k0 < DDIM; k0 += 32) {
        v16bf b[8];
        #pragma unroll
        for (int j = 0; j < 8; ++j)                 // distinct regs -> partial waits
            b[j] = *(const v16bf*)(wcol + (size_t)j * 16 * DDIM + k0);
        const int kn = (k0 + 32 < DDIM) ? (k0 + 32) : 0;   // uniform, clamped
        v16bf an = load_a(xrow + kn + hi * 16);            // pipeline next A
        #pragma unroll
        for (int j = 0; j < 8; ++j)
            acc[j] = __builtin_amdgcn_wmma_f32_16x16x32_bf16(
                         false, a, false, b[j], (short)0, acc[j], false, false);
        a = an;
    }

    #pragma unroll
    for (int j = 0; j < 8; ++j) {
        const int o = oBase + j * 16 + m;        // C/D col = lane%16
        const float bias = bs[o];
        #pragma unroll
        for (int r = 0; r < 8; ++r) {
            const int row = nBase + r + 8 * hi;  // C/D: lanes 16-31 hold M=8..15
            out[(size_t)row * DDIM + o] = acc[j][r] + bias;
        }
    }
}

// ---------------------------------------------------------------------------
// Kernel 3: routed experts, gathered top-2 only (~2N rows instead of 16N).
// slots[tok][slot][o] = w * (x[tok] . Wr[e][o] + br[e][o])
// ---------------------------------------------------------------------------
__global__ void __launch_bounds__(256)
routed_gemm_kernel(const __bf16* __restrict__ xb, const __bf16* __restrict__ Wrb,
                   const float* __restrict__ br, const int* __restrict__ cnt,
                   const int* __restrict__ list_tok, const float* __restrict__ list_w,
                   float* __restrict__ slots)
{
    const int e     = blockIdx.y;
    const int lane  = threadIdx.x & 31;
    const int wave  = threadIdx.x >> 5;
    const int chunk = (blockIdx.x >> 2) * 8 + wave;           // 0..1023
    const int oBase = (blockIdx.x & 3) * 128;
    const int cnt_e = cnt[e];
    if (chunk * 16 >= cnt_e) return;                          // wave-uniform exit

    const int m  = lane & 15;
    const int hi = lane >> 4;
    const int lbase = e * N_TOK + chunk * 16;

    int tokA = 0;
    if (chunk * 16 + m < cnt_e) tokA = list_tok[lbase + m] & 0xFFFFF;  // value select only
    const __bf16* xrow = xb + (size_t)tokA * DDIM;
    const __bf16* wcol = Wrb + (size_t)e * DDIM * DDIM
                             + (size_t)(oBase + m) * DDIM + hi * 16;

    v8f acc[8] = {};
    v16bf a = load_a(xrow + hi * 16);
    #pragma unroll 1
    for (int k0 = 0; k0 < DDIM; k0 += 32) {
        v16bf b[8];
        #pragma unroll
        for (int j = 0; j < 8; ++j)
            b[j] = *(const v16bf*)(wcol + (size_t)j * 16 * DDIM + k0);
        const int kn = (k0 + 32 < DDIM) ? (k0 + 32) : 0;
        v16bf an = load_a(xrow + kn + hi * 16);
        #pragma unroll
        for (int j = 0; j < 8; ++j)
            acc[j] = __builtin_amdgcn_wmma_f32_16x16x32_bf16(
                         false, a, false, b[j], (short)0, acc[j], false, false);
        a = an;
    }

    const float* bre = br + e * DDIM;
    #pragma unroll
    for (int r = 0; r < 8; ++r) {
        const int rowIdx = chunk * 16 + r + 8 * hi;
        if (rowIdx < cnt_e) {
            const int   entry = list_tok[e * N_TOK + rowIdx];
            const int   tok   = entry & 0xFFFFF;
            const int   slot  = entry >> 20;
            const float w     = list_w[e * N_TOK + rowIdx];
            #pragma unroll
            for (int j = 0; j < 8; ++j) {
                const int o = oBase + j * 16 + m;
                slots[(size_t)tok * (2 * DDIM) + slot * DDIM + o] = w * (acc[j][r] + bre[o]);
            }
        }
    }
}

// ---------------------------------------------------------------------------
// Kernel 4: out = relu(shared + x + slot0 + slot1), float4 vectorized.
// ---------------------------------------------------------------------------
__global__ void __launch_bounds__(256)
combine_kernel(const float* __restrict__ x, const float* __restrict__ slots,
               float* __restrict__ out)
{
    const int t   = blockIdx.x * 256 + threadIdx.x;
    const int idx = t * 4;
    const int n   = idx >> 9;
    const int d   = idx & 511;
    f32x4 s  = *(const f32x4*)(out + idx);
    f32x4 xv = *(const f32x4*)(x + idx);
    f32x4 s0 = *(const f32x4*)(slots + (size_t)n * 1024 + d);
    f32x4 s1 = *(const f32x4*)(slots + (size_t)n * 1024 + 512 + d);
    f32x4 r;
    #pragma unroll
    for (int i = 0; i < 4; ++i) {
        float v = s[i] + xv[i] + s0[i] + s1[i];
        r[i] = v > 0.0f ? v : 0.0f;
    }
    *(f32x4*)(out + idx) = r;
}

extern "C" void kernel_launch(void* const* d_in, const int* in_sizes, int n_in,
                              void* d_out, int out_size, void* d_ws, size_t ws_size,
                              hipStream_t stream)
{
    const float* x  = (const float*)d_in[0];
    const float* Ws = (const float*)d_in[1];
    const float* bs = (const float*)d_in[2];
    const float* Wr = (const float*)d_in[3];
    const float* br = (const float*)d_in[4];
    const float* Wg = (const float*)d_in[5];
    const float* bg = (const float*)d_in[6];
    const float* gb = (const float*)d_in[7];
    float* out = (float*)d_out;

    // Workspace layout (all offsets 256B aligned):
    //   cnt[16] | list_tok[E*N] | list_w[E*N] | slots[N][2][D] f32
    //   | x_bf[N][D] | Ws_bf[D][D] | Wr_bf[E][D][D]
    char*  ws       = (char*)d_ws;
    size_t off      = 0;
    int*   cnt      = (int*)(ws + off);   off += 256;
    int*   list_tok = (int*)(ws + off);   off += (size_t)NEXP * N_TOK * 4;
    float* list_w   = (float*)(ws + off); off += (size_t)NEXP * N_TOK * 4;
    float* slots    = (float*)(ws + off); off += (size_t)N_TOK * 2 * DDIM * 4;
    __bf16* x_bf    = (__bf16*)(ws + off); off += (size_t)N_TOK * DDIM * 2;
    __bf16* Ws_bf   = (__bf16*)(ws + off); off += (size_t)DDIM * DDIM * 2;
    __bf16* Wr_bf   = (__bf16*)(ws + off); off += (size_t)NEXP * DDIM * DDIM * 2;

    hipMemsetAsync(cnt, 0, NEXP * sizeof(int), stream);   // graph-capturable

    // one-pass f32 -> bf16 of x, Ws, Wr (all counts divisible by 8*256)
    convert_bf16_kernel<<<(N_TOK * DDIM) / 8 / 256, 256, 0, stream>>>(x, x_bf,
                                                        (N_TOK * DDIM) / 8);
    convert_bf16_kernel<<<(DDIM * DDIM) / 8 / 256, 256, 0, stream>>>(Ws, Ws_bf,
                                                        (DDIM * DDIM) / 8);
    convert_bf16_kernel<<<(NEXP * DDIM * DDIM) / 8 / 256, 256, 0, stream>>>(Wr, Wr_bf,
                                                        (NEXP * DDIM * DDIM) / 8);

    gate_kernel<<<N_TOK / 256, 256, 0, stream>>>(x, Wg, bg, gb, cnt, list_tok, list_w);

    // 1024 n-tiles * 4 o-quads = 4096 waves = 512 blocks of 8 waves
    shared_gemm_kernel<<<512, 256, 0, stream>>>(x_bf, Ws_bf, bs, out);

    // worst case 1024 chunks * 4 o-quads per expert; inactive waves exit fast
    routed_gemm_kernel<<<dim3(512, NEXP), 256, 0, stream>>>(x_bf, Wr_bf, br, cnt,
                                                            list_tok, list_w, slots);

    combine_kernel<<<(N_TOK * DDIM) / (4 * 256), 256, 0, stream>>>(x, slots, out);
}
